// GAT_36696200577383
// MI455X (gfx1250) — matrix operand order repaired and checked by reading, not compiled, not run on hardware
//
#include <hip/hip_runtime.h>
#include <stdint.h>

// ---------------------------------------------------------------------------
// GAT (2-layer) forward for MI455X / gfx1250.
//   L1: h1 = x @ W1 (bf16 WMMA, f32 acc) -> per-head attention logits ->
//       edge segment-softmax (float atomic max / add) ->
//       scatter-aggregate (1 block == 1 edge, coalesced 1KB gather+scatter) ->
//       + b1 -> BatchNorm (batch stats) -> ELU
//   L2: h2 = h' @ W2 (bf16 WMMA, N padded 40->48) -> 1-head softmax ->
//       scatter-aggregate into d_out (+ b2).
// ---------------------------------------------------------------------------

typedef __bf16 bf16_t;
typedef bf16_t v16bf __attribute__((ext_vector_type(16)));
typedef float  v8f   __attribute__((ext_vector_type(8)));
typedef unsigned short u16;

union FragBF { uint4 q[2]; v16bf v; };

#define F_HID 256
#define HEADS 8
#define PER_HEAD 32
#define CLS 40
#define CLS_PAD 48

// ---- helpers --------------------------------------------------------------

__device__ __forceinline__ u16 f2bf(float f) {
  unsigned u = __float_as_uint(f);
  unsigned r = (u + 0x7FFFu + ((u >> 16) & 1u)) >> 16;   // RNE
  return (u16)r;
}

__device__ __forceinline__ void atomicMaxF(float* addr, float v) {
  // signed-max for non-negative patterns, unsigned-min for negative patterns
  if (v >= 0.f) atomicMax((int*)addr, __float_as_int(v));
  else          atomicMin((unsigned int*)addr, __float_as_uint(v));
}

__device__ __forceinline__ float leaky02(float v) { return v > 0.f ? v : 0.2f * v; }

// ---- init kernels ---------------------------------------------------------

__global__ void fill_u32(unsigned* p, unsigned v, size_t n) {
  size_t i = (size_t)blockIdx.x * blockDim.x + threadIdx.x;
  size_t stride = (size_t)gridDim.x * blockDim.x;
  for (; i < n; i += stride) p[i] = v;
}

__global__ void init_bias_rows(float* p, const float* __restrict__ bias,
                               int rowlen, size_t total) {
  size_t i = (size_t)blockIdx.x * blockDim.x + threadIdx.x;
  size_t stride = (size_t)gridDim.x * blockDim.x;
  for (; i < total; i += stride) p[i] = bias[i % rowlen];
}

// ---- bf16 conversion + weight transpose -----------------------------------

__global__ void convert_inputs(const float* __restrict__ x,
                               const float* __restrict__ W1,
                               const float* __restrict__ W2,
                               u16* __restrict__ xb,
                               u16* __restrict__ w1t,   // [256 cols][256 k] col-major
                               u16* __restrict__ w2t,   // [48 cols][256 k] col-major, padded
                               int n) {
  size_t i = (size_t)blockIdx.x * blockDim.x + threadIdx.x;
  size_t stride = (size_t)gridDim.x * blockDim.x;
  size_t totX = (size_t)n * F_HID;
  for (size_t t = i; t < totX; t += stride) xb[t] = f2bf(x[t]);
  for (size_t t = i; t < (size_t)F_HID * F_HID; t += stride) {
    int c = (int)(t >> 8), k = (int)(t & 255);
    w1t[t] = f2bf(W1[(size_t)k * F_HID + c]);          // transpose
  }
  for (size_t t = i; t < (size_t)CLS_PAD * F_HID; t += stride) {
    int c = (int)(t >> 8), k = (int)(t & 255);
    w2t[t] = (c < CLS) ? f2bf(W2[(size_t)k * CLS + c]) : (u16)0;
  }
}

// ---- GEMM1: h1[N,256] = x[N,256] @ W1[256,256]   (bf16 WMMA, f32 acc) -----
// grid.x = row tiles (16 rows each); block = 256 = 8 waves;
// wave w computes column tiles w and w+8.

__global__ __launch_bounds__(256) void gemm1_wmma(const u16* __restrict__ xb,
                                                  const u16* __restrict__ w1t,
                                                  float* __restrict__ h1,
                                                  int n) {
  const int wave = threadIdx.x >> 5;
  const int lane = threadIdx.x & 31;
  const int half = lane >> 4;
  const int m    = lane & 15;

  const int row0 = blockIdx.x * 16;
  int rowA = row0 + m; if (rowA >= n) rowA = n - 1;     // clamp, keep EXEC full
  const int col0 = wave * 16;
  const int col1 = (wave + 8) * 16;

  const u16* arow  = xb  + (size_t)rowA * F_HID;
  const u16* bcol0 = w1t + (size_t)(col0 + m) * F_HID;
  const u16* bcol1 = w1t + (size_t)(col1 + m) * F_HID;

  v8f acc0 = {}; v8f acc1 = {};
#pragma unroll
  for (int k0 = 0; k0 < F_HID; k0 += 32) {
    FragBF a, b0, b1;
    a.q[0]  = *(const uint4*)(arow  + k0 + half * 16);       // K = half*16 .. +7
    a.q[1]  = *(const uint4*)(arow  + k0 + half * 16 + 8);   // K = half*16+8 ..
    b0.q[0] = *(const uint4*)(bcol0 + k0 + half * 16);       // K = half*16 .. +7
    b0.q[1] = *(const uint4*)(bcol0 + k0 + half * 16 + 8);
    b1.q[0] = *(const uint4*)(bcol1 + k0 + half * 16);
    b1.q[1] = *(const uint4*)(bcol1 + k0 + half * 16 + 8);
    acc0 = __builtin_amdgcn_wmma_f32_16x16x32_bf16(false, a.v, false, b0.v,
                                                   (short)0, acc0, false, false);
    acc1 = __builtin_amdgcn_wmma_f32_16x16x32_bf16(false, a.v, false, b1.v,
                                                   (short)0, acc1, false, false);
  }
#pragma unroll
  for (int r = 0; r < 8; ++r) {
    int row = row0 + half * 8 + r;
    if (row < n) {
      h1[(size_t)row * F_HID + col0 + m] = acc0[r];
      h1[(size_t)row * F_HID + col1 + m] = acc1[r];
    }
  }
}

// ---- attention logits, layer 1: one thread per (node, head) ---------------

__global__ void alpha1_kernel(const float* __restrict__ h1,
                              const float* __restrict__ attS,
                              const float* __restrict__ attD,
                              float* __restrict__ as1, float* __restrict__ ad1,
                              int n) {
  int t = blockIdx.x * blockDim.x + threadIdx.x;
  if (t >= n * HEADS) return;
  int node = t >> 3, h = t & 7;
  const float* p  = h1 + (size_t)node * F_HID + h * PER_HEAD;
  const float* as = attS + h * PER_HEAD;
  const float* ad = attD + h * PER_HEAD;
  float ss = 0.f, dd = 0.f;
#pragma unroll
  for (int i = 0; i < PER_HEAD; ++i) { float v = p[i]; ss += v * as[i]; dd += v * ad[i]; }
  as1[t] = ss; ad1[t] = dd;
}

// ---- layer-1 edge segment softmax -----------------------------------------
// E' = E + N (self-loops appended, matching jnp.concatenate order).

__global__ void edge_max1(const int* __restrict__ ei,
                          const float* __restrict__ as1, const float* __restrict__ ad1,
                          float* __restrict__ mx1, int E, int EP) {
  long long t = (long long)blockIdx.x * blockDim.x + threadIdx.x;
  if (t >= (long long)EP * HEADS) return;
  int e = (int)(t >> 3), h = (int)(t & 7);
  int s, d;
  if (e < E) { s = ei[e]; d = ei[E + e]; } else { s = d = e - E; }
  float v = leaky02(as1[s * HEADS + h] + ad1[d * HEADS + h]);
  atomicMaxF(&mx1[d * HEADS + h], v);
}

__global__ void edge_sum1(const int* __restrict__ ei,
                          const float* __restrict__ as1, const float* __restrict__ ad1,
                          const float* __restrict__ mx1, float* __restrict__ sm1,
                          int E, int EP) {
  long long t = (long long)blockIdx.x * blockDim.x + threadIdx.x;
  if (t >= (long long)EP * HEADS) return;
  int e = (int)(t >> 3), h = (int)(t & 7);
  int s, d;
  if (e < E) { s = ei[e]; d = ei[E + e]; } else { s = d = e - E; }
  float v = leaky02(as1[s * HEADS + h] + ad1[d * HEADS + h]);
  atomicAdd(&sm1[d * HEADS + h], __expf(v - mx1[d * HEADS + h]));
}

// one 256-thread block per edge: coalesced 1KB gather + 1KB atomic scatter
__global__ __launch_bounds__(256) void edge_aggregate1(
    const int* __restrict__ ei,
    const float* __restrict__ as1, const float* __restrict__ ad1,
    const float* __restrict__ mx1, const float* __restrict__ sm1,
    const float* __restrict__ h1, float* __restrict__ out1, int E, int EP) {
  int e = blockIdx.x;
  if (e >= EP) return;
  int s, d;
  if (e < E) { s = ei[e]; d = ei[E + e]; } else { s = d = e - E; }
  int c = threadIdx.x;                 // channel 0..255
  int h = c >> 5;                      // head
  const float* srow = h1 + (size_t)s * F_HID;
  __builtin_prefetch(srow + c, 0, 0);  // global_prefetch_b8
  float v = leaky02(as1[s * HEADS + h] + ad1[d * HEADS + h]);
  float alpha = __expf(v - mx1[d * HEADS + h]) / (sm1[d * HEADS + h] + 1e-16f);
  atomicAdd(&out1[(size_t)d * F_HID + c], alpha * srow[c]);
}

// ---- BatchNorm stats + apply + ELU + bf16 cast ----------------------------

__global__ __launch_bounds__(256) void bn_stats(const float* __restrict__ out1,
                                                float* __restrict__ bnsum,
                                                float* __restrict__ bnsq, int n) {
  int c = threadIdx.x;
  float s = 0.f, q = 0.f;
  for (int r = blockIdx.x; r < n; r += gridDim.x) {
    float v = out1[(size_t)r * F_HID + c];
    s += v; q += v * v;
  }
  atomicAdd(&bnsum[c], s);
  atomicAdd(&bnsq[c], q);
}

__global__ void bn_elu(float* __restrict__ out1, u16* __restrict__ h1b,
                       const float* __restrict__ bnsum, const float* __restrict__ bnsq,
                       const float* __restrict__ gamma, const float* __restrict__ beta,
                       float invN, size_t total) {
  size_t i = (size_t)blockIdx.x * blockDim.x + threadIdx.x;
  size_t stride = (size_t)gridDim.x * blockDim.x;
  for (; i < total; i += stride) {
    int c = (int)(i & (F_HID - 1));
    float mu  = bnsum[c] * invN;
    float var = bnsq[c] * invN - mu * mu;
    float v = (out1[i] - mu) * rsqrtf(var + 1e-5f) * gamma[c] + beta[c];
    v = v > 0.f ? v : (__expf(v) - 1.f);                       // ELU
    h1b[i] = f2bf(v);
  }
}

// ---- GEMM2: h2[N,40] = h'[N,256] @ W2[256,40]  (cols padded to 48) --------
// grid.x * 8 waves, each wave owns one 16-row tile, 3 column tiles.

__global__ __launch_bounds__(256) void gemm2_wmma(const u16* __restrict__ h1b,
                                                  const u16* __restrict__ w2t,
                                                  float* __restrict__ h2,
                                                  int n, int nRowTiles) {
  const int wave = threadIdx.x >> 5;
  const int lane = threadIdx.x & 31;
  const int half = lane >> 4;
  const int m    = lane & 15;
  const int rowTile = blockIdx.x * 8 + wave;
  if (rowTile >= nRowTiles) return;                    // wave-uniform exit

  int rowA = rowTile * 16 + m; if (rowA >= n) rowA = n - 1;
  const u16* arow = h1b + (size_t)rowA * F_HID;

  v8f acc[3] = {};
#pragma unroll
  for (int k0 = 0; k0 < F_HID; k0 += 32) {
    FragBF a;
    a.q[0] = *(const uint4*)(arow + k0 + half * 16);
    a.q[1] = *(const uint4*)(arow + k0 + half * 16 + 8);
#pragma unroll
    for (int t = 0; t < 3; ++t) {
      const u16* bcol = w2t + (size_t)(t * 16 + m) * F_HID;
      FragBF b;
      b.q[0] = *(const uint4*)(bcol + k0 + half * 16);
      b.q[1] = *(const uint4*)(bcol + k0 + half * 16 + 8);
      acc[t] = __builtin_amdgcn_wmma_f32_16x16x32_bf16(false, a.v, false, b.v,
                                                       (short)0, acc[t], false, false);
    }
  }
#pragma unroll
  for (int t = 0; t < 3; ++t) {
    int col = t * 16 + m;
    if (col < CLS) {
#pragma unroll
      for (int r = 0; r < 8; ++r) {
        int row = rowTile * 16 + half * 8 + r;
        if (row < n) h2[(size_t)row * CLS + col] = acc[t][r];
      }
    }
  }
}

// ---- layer-2 attention (1 head, 40 ch) ------------------------------------

__global__ void alpha2_kernel(const float* __restrict__ h2,
                              const float* __restrict__ attS, const float* __restrict__ attD,
                              float* __restrict__ as2, float* __restrict__ ad2, int n) {
  int t = blockIdx.x * blockDim.x + threadIdx.x;
  if (t >= n) return;
  const float* p = h2 + (size_t)t * CLS;
  float ss = 0.f, dd = 0.f;
#pragma unroll
  for (int i = 0; i < CLS; ++i) { float v = p[i]; ss += v * attS[i]; dd += v * attD[i]; }
  as2[t] = ss; ad2[t] = dd;
}

__global__ void edge_max2(const int* __restrict__ ei,
                          const float* __restrict__ as2, const float* __restrict__ ad2,
                          float* __restrict__ mx2, int E, int EP) {
  int e = blockIdx.x * blockDim.x + threadIdx.x;
  if (e >= EP) return;
  int s, d;
  if (e < E) { s = ei[e]; d = ei[E + e]; } else { s = d = e - E; }
  atomicMaxF(&mx2[d], leaky02(as2[s] + ad2[d]));
}

__global__ void edge_sum2(const int* __restrict__ ei,
                          const float* __restrict__ as2, const float* __restrict__ ad2,
                          const float* __restrict__ mx2, float* __restrict__ sm2,
                          int E, int EP) {
  int e = blockIdx.x * blockDim.x + threadIdx.x;
  if (e >= EP) return;
  int s, d;
  if (e < E) { s = ei[e]; d = ei[E + e]; } else { s = d = e - E; }
  float v = leaky02(as2[s] + ad2[d]);
  atomicAdd(&sm2[d], __expf(v - mx2[d]));
}

__global__ void edge_aggregate2(const int* __restrict__ ei,
                                const float* __restrict__ as2, const float* __restrict__ ad2,
                                const float* __restrict__ mx2, const float* __restrict__ sm2,
                                const float* __restrict__ h2, float* __restrict__ out,
                                int E, int EP) {
  long long t = (long long)blockIdx.x * blockDim.x + threadIdx.x;
  if (t >= (long long)EP * CLS) return;
  int e = (int)(t / CLS), c = (int)(t % CLS);
  int s, d;
  if (e < E) { s = ei[e]; d = ei[E + e]; } else { s = d = e - E; }
  float v = leaky02(as2[s] + ad2[d]);
  float alpha = __expf(v - mx2[d]) / (sm2[d] + 1e-16f);
  atomicAdd(&out[(size_t)d * CLS + c], alpha * h2[(size_t)s * CLS + c]);
}

// ---------------------------------------------------------------------------

static inline size_t alignUp(size_t v) { return (v + 255) & ~(size_t)255; }

extern "C" void kernel_launch(void* const* d_in, const int* in_sizes, int n_in,
                              void* d_out, int out_size, void* d_ws, size_t ws_size,
                              hipStream_t stream) {
  const float* x     = (const float*)d_in[0];
  const int*   ei    = (const int*)  d_in[1];
  const float* W1    = (const float*)d_in[2];
  const float* attS1 = (const float*)d_in[3];
  const float* attD1 = (const float*)d_in[4];
  const float* b1    = (const float*)d_in[5];
  const float* gamma = (const float*)d_in[6];
  const float* beta  = (const float*)d_in[7];
  const float* W2    = (const float*)d_in[8];
  const float* attS2 = (const float*)d_in[9];
  const float* attD2 = (const float*)d_in[10];
  const float* b2    = (const float*)d_in[11];
  float* out = (float*)d_out;

  const int N  = in_sizes[0] / F_HID;
  const int E  = in_sizes[1] / 2;
  const int EP = E + N;

  // workspace carve-up
  char* w = (char*)d_ws; size_t off = 0;
  auto take = [&](size_t bytes) { void* p = w + off; off = alignUp(off + bytes); return p; };
  u16*   xb    = (u16*)  take((size_t)N * F_HID * 2);
  u16*   w1t   = (u16*)  take((size_t)F_HID * F_HID * 2);
  u16*   w2t   = (u16*)  take((size_t)CLS_PAD * F_HID * 2);
  float* h1    = (float*)take((size_t)N * F_HID * 4);
  float* as1   = (float*)take((size_t)N * HEADS * 4);
  float* ad1   = (float*)take((size_t)N * HEADS * 4);
  float* mx1   = (float*)take((size_t)N * HEADS * 4);
  float* sm1   = (float*)take((size_t)N * HEADS * 4);
  float* out1  = (float*)take((size_t)N * F_HID * 4);
  float* bnsum = (float*)take((size_t)F_HID * 4);
  float* bnsq  = (float*)take((size_t)F_HID * 4);
  u16*   h1b   = (u16*)  take((size_t)N * F_HID * 2);
  float* h2    = (float*)take((size_t)N * CLS * 4);
  float* as2   = (float*)take((size_t)N * 4);
  float* ad2   = (float*)take((size_t)N * 4);
  float* mx2   = (float*)take((size_t)N * 4);
  float* sm2   = (float*)take((size_t)N * 4);

  const unsigned NEG_INF = 0xFF800000u;
  const int TB = 256;
  const int FILL_BLOCKS = 2048;

  // 0) conversions + inits
  convert_inputs<<<4096, TB, 0, stream>>>(x, W1, W2, xb, w1t, w2t, N);
  fill_u32<<<FILL_BLOCKS, TB, 0, stream>>>((unsigned*)mx1, NEG_INF, (size_t)N * HEADS);
  fill_u32<<<FILL_BLOCKS, TB, 0, stream>>>((unsigned*)sm1, 0u, (size_t)N * HEADS);
  fill_u32<<<1, TB, 0, stream>>>((unsigned*)bnsum, 0u, F_HID);
  fill_u32<<<1, TB, 0, stream>>>((unsigned*)bnsq, 0u, F_HID);
  fill_u32<<<FILL_BLOCKS, TB, 0, stream>>>((unsigned*)mx2, NEG_INF, (size_t)N);
  fill_u32<<<FILL_BLOCKS, TB, 0, stream>>>((unsigned*)sm2, 0u, (size_t)N);
  init_bias_rows<<<FILL_BLOCKS, TB, 0, stream>>>(out1, b1, F_HID, (size_t)N * F_HID);
  init_bias_rows<<<FILL_BLOCKS, TB, 0, stream>>>(out, b2, CLS, (size_t)N * CLS);

  // 1) layer-1 GEMM + logits
  int rowTiles1 = (N + 15) / 16;
  gemm1_wmma<<<rowTiles1, TB, 0, stream>>>(xb, w1t, h1, N);
  alpha1_kernel<<<(N * HEADS + TB - 1) / TB, TB, 0, stream>>>(h1, attS1, attD1, as1, ad1, N);

  // 2) layer-1 edge softmax + aggregation
  long long eh = (long long)EP * HEADS;
  int ehBlocks = (int)((eh + TB - 1) / TB);
  edge_max1<<<ehBlocks, TB, 0, stream>>>(ei, as1, ad1, mx1, E, EP);
  edge_sum1<<<ehBlocks, TB, 0, stream>>>(ei, as1, ad1, mx1, sm1, E, EP);
  edge_aggregate1<<<EP, TB, 0, stream>>>(ei, as1, ad1, mx1, sm1, h1, out1, E, EP);

  // 3) BatchNorm + ELU (+ bf16 cast for GEMM2)
  bn_stats<<<512, TB, 0, stream>>>(out1, bnsum, bnsq, N);
  bn_elu<<<FILL_BLOCKS, TB, 0, stream>>>(out1, h1b, bnsum, bnsq, gamma, beta,
                                         1.0f / (float)N, (size_t)N * F_HID);

  // 4) layer-2 GEMM + logits
  int rowTiles2 = (N + 15) / 16;
  gemm2_wmma<<<(rowTiles2 + 7) / 8, TB, 0, stream>>>(h1b, w2t, h2, N, rowTiles2);
  alpha2_kernel<<<(N + TB - 1) / TB, TB, 0, stream>>>(h2, attS2, attD2, as2, ad2, N);

  // 5) layer-2 edge softmax + aggregation into d_out
  int eBlocks = (EP + TB - 1) / TB;
  edge_max2<<<eBlocks, TB, 0, stream>>>(ei, as2, ad2, mx2, E, EP);
  edge_sum2<<<eBlocks, TB, 0, stream>>>(ei, as2, ad2, mx2, sm2, E, EP);
  long long ec = (long long)EP * CLS;
  edge_aggregate2<<<(int)((ec + TB - 1) / TB), TB, 0, stream>>>(ei, as2, ad2, mx2, sm2,
                                                               h2, out, E, EP);
}
